// Block_47004122087942
// MI455X (gfx1250) — matrix operand order repaired and checked by reading, not compiled, not run on hardware
//
#include <hip/hip_runtime.h>
#include <math.h>

// ---------------- problem constants ----------------
#define BATCH 8
#define TSEQ  1024
#define CE    512          // n_embd
#define NH    8            // heads
#define DH    64           // head dim
#define NE    4            // experts
#define DFF   2048         // ffn dim
#define NTOK  (BATCH * TSEQ)   // 8192

typedef __attribute__((ext_vector_type(16))) __bf16 v16bf;
typedef __attribute__((ext_vector_type(2)))  __bf16 v2bf;
typedef __attribute__((ext_vector_type(8)))  float  v8f;
typedef __attribute__((ext_vector_type(4)))  unsigned v4u;
typedef __attribute__((ext_vector_type(8)))  int      v8i;
typedef __attribute__((ext_vector_type(4)))  int      v4i;

// ---------------- bf16 helpers ----------------
__device__ __forceinline__ unsigned short f2bf(float f) {
    union { float f; unsigned u; } a; a.f = f;
    unsigned u = a.u;
    unsigned r = u + 0x7FFFu + ((u >> 16) & 1u);   // round-to-nearest-even
    return (unsigned short)(r >> 16);
}
__device__ __forceinline__ float bf2f(unsigned short h) {
    union { unsigned u; float f; } a; a.u = ((unsigned)h) << 16;
    return a.f;
}
__device__ __forceinline__ v2bf u2bf2(unsigned int u) {
    union { unsigned u; v2bf v; } c; c.u = u; return c.v;
}
__device__ __forceinline__ __bf16 us2bf(unsigned short u) {
    union { unsigned short s; __bf16 b; } c; c.s = u; return c.b;
}

// ---------------- TDM: 2D bf16 tile -> LDS (ISA 08_async_tensor D# layout) ---
// Loads `rows` rows of `rowElems` bf16 elements, row stride `strideElems`,
// packed contiguously into LDS at byte offset lds_off.
__device__ __forceinline__ void tdm_load_tile_bf16(
        const unsigned short* gptr, unsigned lds_off,
        unsigned rowElems, unsigned rows, unsigned strideElems) {
    unsigned long long ga = (unsigned long long)(size_t)gptr;
    v4u g0;
    g0[0] = 1u;                                   // count=1, user descriptor
    g0[1] = lds_off;                              // LDS byte address
    g0[2] = (unsigned)(ga & 0xffffffffu);         // global addr [31:0]
    g0[3] = (unsigned)((ga >> 32) & 0x01ffffffu)  // global addr [56:32]
            | 0x80000000u;                        // type = 2 ("image")
    unsigned d0 = strideElems;                    // tensor_dim0 (elements)
    unsigned d1 = 0x00100000u;                    // tensor_dim1 (large, in-bounds)
    v8i g1;
    g1[0] = (int)0x00010000u;                     // wg_mask=0, data_size=1 (2B)
    g1[1] = (int)((d0 & 0xffffu) << 16);          // [63:48] = dim0[15:0]
    g1[2] = (int)((d0 >> 16) | ((d1 & 0xffffu) << 16)); // dim0 hi | dim1 lo
    g1[3] = (int)((d1 >> 16) | (rowElems << 16)); // dim1 hi | tile_dim0
    g1[4] = (int)(rows & 0xffffu);                // tile_dim1 ; tile_dim2 = 0
    g1[5] = (int)strideElems;                     // dim0_stride [31:0]
    g1[6] = 0;                                    // dim0_stride hi, dim1_stride lo
    g1[7] = 0;
    v4i z4 = {0, 0, 0, 0};
#if __clang_major__ >= 23
    v8i z8 = {0, 0, 0, 0, 0, 0, 0, 0};
    __builtin_amdgcn_tensor_load_to_lds(g0, g1, z4, z4, z8, 0);
#else
    __builtin_amdgcn_tensor_load_to_lds(g0, g1, z4, z4, 0);
#endif
}
__device__ __forceinline__ unsigned lds_offset_of(const void* shared_ptr) {
    return (unsigned)(size_t)shared_ptr;          // low 32 bits = LDS address
}

// ---------------- fp32 -> bf16 convert ----------------
__global__ void cvt_f32_bf16(const float* __restrict__ src,
                             unsigned short* __restrict__ dst, int n) {
    for (int i = blockIdx.x * blockDim.x + threadIdx.x; i < n;
         i += gridDim.x * blockDim.x)
        dst[i] = f2bf(src[i]);
}

// ---------------- RMSNorm (row per block), bf16 out ----------------
__global__ __launch_bounds__(128)
void rmsnorm_bf16(const float* __restrict__ x, const float* __restrict__ w,
                  unsigned short* __restrict__ out) {
    const int row = blockIdx.x;
    const float* xr = x + (size_t)row * CE;
    float ss = 0.f;
    for (int c = threadIdx.x; c < CE; c += 128) { float v = xr[c]; ss += v * v; }
    for (int m = 1; m < 32; m <<= 1) ss += __shfl_xor(ss, m, 32);
    __shared__ float part[4];
    if ((threadIdx.x & 31) == 0) part[threadIdx.x >> 5] = ss;
    __syncthreads();
    float tot = part[0] + part[1] + part[2] + part[3];
    float scale = rsqrtf(tot / (float)CE + 1e-5f);
    for (int c = threadIdx.x; c < CE; c += 128)
        out[(size_t)row * CE + c] = f2bf(xr[c] * scale * w[c]);
}

// ---------------- Gate: softmax + top-2 renorm -> router [NTOK,4] ----------------
__global__ __launch_bounds__(128)
void gate_kernel(const unsigned short* __restrict__ xn,
                 const float* __restrict__ gw,
                 float* __restrict__ router) {
    const int w = threadIdx.x >> 5, lane = threadIdx.x & 31;
    const int n = blockIdx.x * 4 + w;
    float a0 = 0, a1 = 0, a2 = 0, a3 = 0;
    for (int c = lane; c < CE; c += 32) {
        float xv = bf2f(xn[(size_t)n * CE + c]);
        a0 += xv * gw[0 * CE + c];
        a1 += xv * gw[1 * CE + c];
        a2 += xv * gw[2 * CE + c];
        a3 += xv * gw[3 * CE + c];
    }
    for (int m = 1; m < 32; m <<= 1) {
        a0 += __shfl_xor(a0, m, 32); a1 += __shfl_xor(a1, m, 32);
        a2 += __shfl_xor(a2, m, 32); a3 += __shfl_xor(a3, m, 32);
    }
    if (lane == 0) {
        float l[4] = {a0, a1, a2, a3};
        float mx = fmaxf(fmaxf(l[0], l[1]), fmaxf(l[2], l[3]));
        float p[4], s = 0.f;
        for (int e = 0; e < 4; ++e) { p[e] = __expf(l[e] - mx); s += p[e]; }
        for (int e = 0; e < 4; ++e) p[e] /= s;
        int i1 = 0;
        for (int e = 1; e < 4; ++e) if (p[e] > p[i1]) i1 = e;
        int i2 = (i1 == 0) ? 1 : 0;
        for (int e = 0; e < 4; ++e) if (e != i1 && p[e] > p[i2]) i2 = e;
        float den = fmaxf(p[i1] + p[i2], 1e-6f);
        float o[4] = {0.f, 0.f, 0.f, 0.f};
        o[i1] = p[i1] / den; o[i2] = p[i2] / den;
        for (int e = 0; e < 4; ++e) router[(size_t)n * 4 + e] = o[e];
    }
}

// ---------------- bf16 WMMA GEMM, 128x128 tile, TDM double-buffered ----------
// C[M,N] = A[M,K] * B[N,K]^T
// mode 0: outF = acc (+resid)          mode 1: outH = bf16(gelu(acc))
// mode 2: outH = bf16(acc)             mode 3: outF += rowScale[m*stride] * acc
__global__ __launch_bounds__(256)
void gemm_bf16_ABt(const unsigned short* __restrict__ A,
                   const unsigned short* __restrict__ B,
                   int M, int N, int K,
                   float* __restrict__ outF,
                   unsigned short* __restrict__ outH,
                   const float* __restrict__ resid,
                   const float* __restrict__ rowScale, int rsStride,
                   int mode) {
    __shared__ unsigned short As[2][128 * 32];
    __shared__ unsigned short Bs[2][128 * 32];
    const int tid  = threadIdx.x;
    const int lane = tid & 31;
    const int w    = tid >> 5;           // 8 waves
    const int wm   = (w >> 1) * 32;      // 4 row groups of 32
    const int wn   = (w & 1) * 64;       // 2 col groups of 64
    const int bm   = blockIdx.y * 128;
    const int bn   = blockIdx.x * 128;
    const int ml   = lane & 15;
    const int kh   = lane >> 4;

    const unsigned asOff = lds_offset_of(&As[0][0]);
    const unsigned bsOff = lds_offset_of(&Bs[0][0]);
    const int nk = K >> 5;               // K / 32

    v8f acc[2][4] = {};

    // prefetch k-tile 0 into buffer 0
    if (w == 0) {
        tdm_load_tile_bf16(A + (size_t)bm * K, asOff, 32, 128, (unsigned)K);
        tdm_load_tile_bf16(B + (size_t)bn * K, bsOff, 32, 128, (unsigned)K);
    }
    __builtin_amdgcn_s_wait_tensorcnt(0);
    __syncthreads();

    for (int kt = 0; kt < nk; ++kt) {
        const int cur = kt & 1;
        if (w == 0 && (kt + 1) < nk) {   // async prefetch of next k-tile
            const unsigned nb = (unsigned)((kt + 1) & 1) * (128 * 32 * 2);
            const int k1 = (kt + 1) << 5;
            tdm_load_tile_bf16(A + (size_t)bm * K + k1, asOff + nb, 32, 128, (unsigned)K);
            tdm_load_tile_bf16(B + (size_t)bn * K + k1, bsOff + nb, 32, 128, (unsigned)K);
        }

        v16bf af[2], bfr[4];
        for (int mi = 0; mi < 2; ++mi) {
            int m = wm + mi * 16 + ml;
            v16bf f;
            for (int v = 0; v < 8; ++v) {
                int kk = (v >> 2) * 16 + kh * 8 + (v & 3) * 2;   // A layout
                v2bf p = u2bf2(*(const unsigned int*)&As[cur][m * 32 + kk]);
                f[2 * v] = p[0]; f[2 * v + 1] = p[1];
            }
            af[mi] = f;
        }
        for (int ni = 0; ni < 4; ++ni) {
            int n = wn + ni * 16 + ml;
            v16bf f;
            for (int v = 0; v < 8; ++v) {
                int kk = (v >> 2) * 16 + kh * 8 + (v & 3) * 2;   // B layout
                v2bf p = u2bf2(*(const unsigned int*)&Bs[cur][n * 32 + kk]);
                f[2 * v] = p[0]; f[2 * v + 1] = p[1];
            }
            bfr[ni] = f;
        }
        for (int mi = 0; mi < 2; ++mi)
            for (int ni = 0; ni < 4; ++ni)
                acc[mi][ni] = __builtin_amdgcn_wmma_f32_16x16x32_bf16(
                    false, af[mi], false, bfr[ni], (short)0, acc[mi][ni],
                    false, false);

        __builtin_amdgcn_s_wait_tensorcnt(0);   // next tile landed (wave 0)
        __syncthreads();
    }

    for (int mi = 0; mi < 2; ++mi)
        for (int ni = 0; ni < 4; ++ni)
            for (int i = 0; i < 8; ++i) {
                int m = bm + wm + mi * 16 + i + 8 * kh;
                int n = bn + wn + ni * 16 + ml;
                float val = acc[mi][ni][i];
                size_t idx = (size_t)m * N + n;
                if (mode == 0) {
                    float r = resid ? resid[idx] : 0.f;
                    outF[idx] = val + r;
                } else if (mode == 1) {
                    float g = 0.5f * val * (1.0f + erff(val * 0.70710678118f));
                    outH[idx] = f2bf(g);
                } else if (mode == 2) {
                    outH[idx] = f2bf(val);
                } else {
                    outF[idx] += rowScale[(size_t)m * rsStride] * val;
                }
            }
}

// ---------------- flash attention: one block per (b*h, 64-query tile) --------
__global__ __launch_bounds__(128)
void attn_flash(const unsigned short* __restrict__ Qm,
                const unsigned short* __restrict__ Km,
                const unsigned short* __restrict__ Vm,
                unsigned short* __restrict__ Om) {
    __shared__ unsigned short Qs[64 * 64];
    __shared__ unsigned short Ks[64 * 64];
    __shared__ unsigned short Vs[64 * 64];
    __shared__ unsigned short Ps[64 * 64];

    const int tid  = threadIdx.x;
    const int lane = tid & 31;
    const int w    = tid >> 5;
    const int b    = blockIdx.x / NH;
    const int h    = blockIdx.x % NH;
    const int qb   = blockIdx.y;
    const int ml   = lane & 15;
    const int kh   = lane >> 4;
    const int m0   = w * 16;

    // Q tile via TDM
    if (w == 0)
        tdm_load_tile_bf16(Qm + (size_t)(b * TSEQ + qb * 64) * CE + h * DH,
                           lds_offset_of(&Qs[0]), 64, 64, CE);
    __builtin_amdgcn_s_wait_tensorcnt(0);
    __syncthreads();

    v16bf aQ[2];
    for (int s = 0; s < 2; ++s) {
        v16bf f;
        for (int v = 0; v < 8; ++v) {
            int k = s * 32 + (v >> 2) * 16 + kh * 8 + (v & 3) * 2;
            v2bf p = u2bf2(*(const unsigned int*)&Qs[(m0 + ml) * 64 + k]);
            f[2 * v] = p[0]; f[2 * v + 1] = p[1];
        }
        aQ[s] = f;
    }

    float Mrow[8], Lrow[8];
    v8f Oc[4] = {};
    for (int i = 0; i < 8; ++i) { Mrow[i] = -1e30f; Lrow[i] = 0.f; }

    for (int jb = 0; jb <= qb; ++jb) {
        __syncthreads();                 // everyone done with previous K/V tile
        if (w == 0) {
            const unsigned short* kbase = Km + (size_t)(b * TSEQ + jb * 64) * CE + h * DH;
            const unsigned short* vbase = Vm + (size_t)(b * TSEQ + jb * 64) * CE + h * DH;
            tdm_load_tile_bf16(kbase, lds_offset_of(&Ks[0]), 64, 64, CE);
            tdm_load_tile_bf16(vbase, lds_offset_of(&Vs[0]), 64, 64, CE);
        }
        __builtin_amdgcn_s_wait_tensorcnt(0);
        __syncthreads();

        v8f S[4] = {};
        for (int nt = 0; nt < 4; ++nt)
            for (int s = 0; s < 2; ++s) {
                v16bf f;
                int n = nt * 16 + ml;
                for (int v = 0; v < 8; ++v) {
                    int kk = s * 32 + (v >> 2) * 16 + kh * 8 + (v & 3) * 2;
                    v2bf p = u2bf2(*(const unsigned int*)&Ks[n * 64 + kk]);
                    f[2 * v] = p[0]; f[2 * v + 1] = p[1];
                }
                S[nt] = __builtin_amdgcn_wmma_f32_16x16x32_bf16(
                    false, aQ[s], false, f, (short)0, S[nt], false, false);
            }

        // scale + causal mask
        for (int nt = 0; nt < 4; ++nt)
            for (int i = 0; i < 8; ++i) {
                float sv = S[nt][i] * 0.125f;   // 1/sqrt(64)
                int kg = jb * 64 + nt * 16 + ml;
                int qg = qb * 64 + m0 + i + 8 * kh;
                S[nt][i] = (kg > qg) ? -1e30f : sv;
            }

        // online softmax
        float alpha[8];
        for (int i = 0; i < 8; ++i) {
            float rm = fmaxf(fmaxf(S[0][i], S[1][i]), fmaxf(S[2][i], S[3][i]));
            for (int msk = 1; msk < 16; msk <<= 1)
                rm = fmaxf(rm, __shfl_xor(rm, msk, 32));
            float newM = fmaxf(Mrow[i], rm);
            alpha[i] = __expf(Mrow[i] - newM);
            Mrow[i] = newM;
        }
        for (int i = 0; i < 8; ++i) {
            float rs = 0.f;
            int prow = m0 + i + 8 * kh;
            for (int nt = 0; nt < 4; ++nt) {
                float p = __expf(S[nt][i] - Mrow[i]);
                rs += p;
                Ps[prow * 64 + nt * 16 + ml] = f2bf(p);
            }
            for (int msk = 1; msk < 16; msk <<= 1) rs += __shfl_xor(rs, msk, 32);
            Lrow[i] = Lrow[i] * alpha[i] + rs;
        }
        for (int nt = 0; nt < 4; ++nt)
            for (int i = 0; i < 8; ++i) Oc[nt][i] *= alpha[i];

        // P @ V
        v16bf aP[2];
        for (int s = 0; s < 2; ++s) {
            v16bf f;
            for (int v = 0; v < 8; ++v) {
                int k = s * 32 + (v >> 2) * 16 + kh * 8 + (v & 3) * 2;
                v2bf p = u2bf2(*(const unsigned int*)&Ps[(m0 + ml) * 64 + k]);
                f[2 * v] = p[0]; f[2 * v + 1] = p[1];
            }
            aP[s] = f;
        }
        for (int nt = 0; nt < 4; ++nt)
            for (int s = 0; s < 2; ++s) {
                v16bf f;
                for (int v = 0; v < 8; ++v) {
                    int kk = s * 32 + (v >> 2) * 16 + kh * 8 + (v & 3) * 2;
                    f[2 * v]     = us2bf(Vs[kk * 64 + nt * 16 + ml]);
                    f[2 * v + 1] = us2bf(Vs[(kk + 1) * 64 + nt * 16 + ml]);
                }
                Oc[nt] = __builtin_amdgcn_wmma_f32_16x16x32_bf16(
                    false, aP[s], false, f, (short)0, Oc[nt], false, false);
            }
    }

    for (int nt = 0; nt < 4; ++nt)
        for (int i = 0; i < 8; ++i) {
            int row = b * TSEQ + qb * 64 + m0 + i + 8 * kh;
            int col = h * DH + nt * 16 + ml;
            Om[(size_t)row * CE + col] = f2bf(Oc[nt][i] / Lrow[i]);
        }
}

// ---------------- host side ----------------
extern "C" void kernel_launch(void* const* d_in, const int* in_sizes, int n_in,
                              void* d_out, int out_size, void* d_ws, size_t ws_size,
                              hipStream_t stream) {
    const float* x   = (const float*)d_in[0];
    const float* ln1 = (const float*)d_in[1];
    const float* ln2 = (const float*)d_in[2];
    const float* wq  = (const float*)d_in[3];
    const float* wk  = (const float*)d_in[4];
    const float* wv  = (const float*)d_in[5];
    const float* wo  = (const float*)d_in[6];
    const float* gw  = (const float*)d_in[7];
    const float* wfc = (const float*)d_in[8];
    const float* wpr = (const float*)d_in[9];

    float* outX   = (float*)d_out;                       // [NTOK, CE]
    float* router = outX + (size_t)NTOK * CE;            // [NTOK, NE]

    size_t off = 0;
    auto carve = [&](size_t elems) {
        void* p = (char*)d_ws + off;
        off += (elems * sizeof(unsigned short) + 255) & ~(size_t)255;
        return (unsigned short*)p;
    };
    unsigned short* wq_h  = carve((size_t)CE * CE);
    unsigned short* wk_h  = carve((size_t)CE * CE);
    unsigned short* wv_h  = carve((size_t)CE * CE);
    unsigned short* wo_h  = carve((size_t)CE * CE);
    unsigned short* wfc_h = carve((size_t)NE * DFF * CE);
    unsigned short* wpr_h = carve((size_t)NE * CE * DFF);
    unsigned short* xn_h  = carve((size_t)NTOK * CE);
    unsigned short* q_h   = carve((size_t)NTOK * CE);
    unsigned short* k_h   = carve((size_t)NTOK * CE);
    unsigned short* v_h   = carve((size_t)NTOK * CE);
    unsigned short* ao_h  = carve((size_t)NTOK * CE);
    unsigned short* xn2_h = carve((size_t)NTOK * CE);
    unsigned short* h_h   = carve((size_t)NTOK * DFF);

    // 1) weights -> bf16
    cvt_f32_bf16<<<dim3(512), dim3(256), 0, stream>>>(wq, wq_h, CE * CE);
    cvt_f32_bf16<<<dim3(512), dim3(256), 0, stream>>>(wk, wk_h, CE * CE);
    cvt_f32_bf16<<<dim3(512), dim3(256), 0, stream>>>(wv, wv_h, CE * CE);
    cvt_f32_bf16<<<dim3(512), dim3(256), 0, stream>>>(wo, wo_h, CE * CE);
    cvt_f32_bf16<<<dim3(2048), dim3(256), 0, stream>>>(wfc, wfc_h, NE * DFF * CE);
    cvt_f32_bf16<<<dim3(2048), dim3(256), 0, stream>>>(wpr, wpr_h, NE * CE * DFF);

    // 2) xn = rmsnorm(x, ln1)
    rmsnorm_bf16<<<dim3(NTOK), dim3(128), 0, stream>>>(x, ln1, xn_h);

    // 3) Q/K/V projections
    dim3 gP(CE / 128, NTOK / 128), blk(256);
    gemm_bf16_ABt<<<gP, blk, 0, stream>>>(xn_h, wq_h, NTOK, CE, CE,
                                          nullptr, q_h, nullptr, nullptr, 0, 2);
    gemm_bf16_ABt<<<gP, blk, 0, stream>>>(xn_h, wk_h, NTOK, CE, CE,
                                          nullptr, k_h, nullptr, nullptr, 0, 2);
    gemm_bf16_ABt<<<gP, blk, 0, stream>>>(xn_h, wv_h, NTOK, CE, CE,
                                          nullptr, v_h, nullptr, nullptr, 0, 2);

    // 4) flash attention
    attn_flash<<<dim3(BATCH * NH, TSEQ / 64), dim3(128), 0, stream>>>(q_h, k_h, v_h, ao_h);

    // 5) out projection + residual: outX = x + ao @ wo^T
    gemm_bf16_ABt<<<gP, blk, 0, stream>>>(ao_h, wo_h, NTOK, CE, CE,
                                          outX, nullptr, x, nullptr, 0, 0);

    // 6) xn2 = rmsnorm(outX, ln2)
    rmsnorm_bf16<<<dim3(NTOK), dim3(128), 0, stream>>>(outX, ln2, xn2_h);

    // 7) gate -> router weights
    gate_kernel<<<dim3(NTOK / 4), dim3(128), 0, stream>>>(xn2_h, gw, router);

    // 8) MoE: dense expert compute, weighted accumulate onto residual in outX
    dim3 gFC(DFF / 128, NTOK / 128);
    for (int e = 0; e < NE; ++e) {
        gemm_bf16_ABt<<<gFC, blk, 0, stream>>>(
            xn2_h, wfc_h + (size_t)e * DFF * CE, NTOK, DFF, CE,
            nullptr, h_h, nullptr, nullptr, 0, 1);                 // gelu -> bf16
        gemm_bf16_ABt<<<gP, blk, 0, stream>>>(
            h_h, wpr_h + (size_t)e * CE * DFF, NTOK, CE, DFF,
            outX, nullptr, nullptr, router + e, NE, 3);            // += w_e * eo
    }
}
